// Reservoir_79946521247760
// MI455X (gfx1250) — matrix operand order repaired and checked by reading, not compiled, not run on hardware
//
#include <hip/hip_runtime.h>
#include <hip/hip_bf16.h>
#include <math.h>

// ---------------------------------------------------------------------------
// Echo-state reservoir on MI455X (gfx1250, wave32, WMMA).
//
// Strategy: W (2048x2048) is converted to bf16 and made LDS-RESIDENT across
// 32 workgroups (64 rows x 2048 cols x 2B = 256 KB < 320 KB/WGP). The 4095
// sequential steps then run entirely out of LDS via v_wmma_f32_16x16x32_bf16
// (f32 accumulation), with only an 8 KB state broadcast through L2 per step
// and a 32-WG monotonic atomic barrier for the serial dependency.
// ---------------------------------------------------------------------------

typedef __bf16 v8bf  __attribute__((ext_vector_type(8)));
typedef __bf16 v16bf __attribute__((ext_vector_type(16)));
typedef float  v8f   __attribute__((ext_vector_type(8)));

#define R           2048
#define T_STEPS     4096
#define N_IN        8
#define N_OUT       4
#define NWG         32
#define ROWS_PER_WG 64            // 2048 / 32
#define BLOCK       128           // 4 wave32 per WG, 16 rows per wave
#define WAVES       4
#define KPAD        8             // pad LDS row stride: 2056*2B = 4112B -> lanes
                                  // stride 4 banks apart => all 64 banks hit
#define WSTRIDE     (R + KPAD)
#define LDS_W_ELEMS (ROWS_PER_WG * WSTRIDE)
#define LDS_BYTES   (LDS_W_ELEMS * 2 + R * 2 + WAVES * 16 * 4 + 64)

__global__ void reservoir_kernel(const float* __restrict__ inputs,
                                 const float* __restrict__ outputs,
                                 const float* __restrict__ w,
                                 const float* __restrict__ w_in,
                                 const float* __restrict__ w_feedb,
                                 const float* __restrict__ noise,
                                 float*       __restrict__ states,     // [T, R]
                                 unsigned*    __restrict__ barrier_cnt)
{
    extern __shared__ char smem[];
    __bf16* Wt   = (__bf16*)smem;                                  // 64 x 2056
    __bf16* xb   = Wt + LDS_W_ELEMS;                               // x_{n-1}, bf16
    float*  ybuf = (float*)(smem + (size_t)LDS_W_ELEMS * 2 + R * 2); // 4 waves x 16

    const int tid     = threadIdx.x;
    const int lane    = tid & 31;
    const int wave    = tid >> 5;
    const int rowbase = blockIdx.x * ROWS_PER_WG;

    // ---- one-time stage: this WG's 64 rows of W, f32 -> bf16, into LDS ----
    for (int i = tid; i < ROWS_PER_WG * R; i += BLOCK) {
        int rr = i >> 11;            // / 2048
        int cc = i & (R - 1);
        Wt[rr * WSTRIDE + cc] = (__bf16)w[(size_t)(rowbase + rr) * R + cc];
    }
    for (int i = tid; i < R; i += BLOCK) xb[i] = (__bf16)0.0f;     // x_0 = 0
    for (int i = tid; i < ROWS_PER_WG; i += BLOCK)                 // states[0] = 0
        states[rowbase + i] = 0.0f;
    __syncthreads();

    // A-fragment addressing (16-bit A 16x32; contiguous k order used here, same as B):
    //   lane L holds row M = L%16; K in [bkb, bkb+16), bkb = L<16?0:16
    // B-fragment (32x16, x replicated into all 16 columns):
    //   lane L holds col N = L%16; K in [bkb, bkb+16), bkb = L<16?0:16
    const int mrow = lane & 15;
    const int kb   = (lane < 16) ? 0 : 8;
    const int bkb  = (lane < 16) ? 0 : 16;
    const int wrow = wave * 16;

    for (int n = 1; n < T_STEPS; ++n) {
        // prefetch next step's noise slice for this WG (global_prefetch_b8)
        if (n + 1 < T_STEPS && tid < ROWS_PER_WG)
            __builtin_prefetch(noise + (size_t)(n + 1) * R + rowbase + tid, 0, 0);

        // ---- y[16 rows] = W_tile * x  via 64 chained bf16 WMMAs ----
        v8f acc = {0.f, 0.f, 0.f, 0.f, 0.f, 0.f, 0.f, 0.f};
        const __bf16* arow = Wt + (size_t)(wrow + mrow) * WSTRIDE;
        #pragma unroll 4
        for (int kc = 0; kc < R; kc += 32) {
            v8bf a_lo = *(const v8bf*)(arow + kc + bkb);
            v8bf a_hi = *(const v8bf*)(arow + kc + bkb + 8);
            v16bf a;
            #pragma unroll
            for (int e = 0; e < 8; ++e) { a[e] = a_lo[e]; a[8 + e] = a_hi[e]; }
            v16bf b = *(const v16bf*)(xb + kc + bkb);   // half-wave broadcast read
            acc = __builtin_amdgcn_wmma_f32_16x16x32_bf16(
                      false, a, false, b, (short)0, acc, false, false);
        }

        // All 16 columns of D are identical: lanes 0/16 hold rows 0..7 / 8..15
        // in acc[0..7]. Spread the 16 row sums across 16 lanes via LDS.
        if (lane == 0 || lane == 16) {
            int mb = (lane == 0) ? 0 : 8;
            #pragma unroll
            for (int j = 0; j < 8; ++j) ybuf[wave * 16 + mb + j] = acc[j];
        }
        __syncthreads();

        if (lane < 16) {
            int r = rowbase + wrow + lane;
            float y = ybuf[wave * 16 + lane];
            const float* u  = inputs  + (size_t)n * N_IN;
            const float* yp = outputs + (size_t)(n - 1) * N_OUT;
            float bsum = 0.f;
            #pragma unroll
            for (int i = 0; i < N_IN;  ++i) bsum += w_in  [(size_t)r * N_IN  + i] * u[i];
            #pragma unroll
            for (int i = 0; i < N_OUT; ++i) bsum += w_feedb[(size_t)r * N_OUT + i] * yp[i];
            float xn = tanhf(y + bsum) + noise[(size_t)n * R + r];
            states[(size_t)n * R + r] = xn;
        }

        // ---- device-wide sync: monotonic counter, release/acquire in L2 ----
        __syncthreads();
        if (tid == 0) {
            __hip_atomic_fetch_add(barrier_cnt, 1u, __ATOMIC_RELEASE,
                                   __HIP_MEMORY_SCOPE_AGENT);
            const unsigned target = (unsigned)NWG * (unsigned)n;
            while (__hip_atomic_load(barrier_cnt, __ATOMIC_ACQUIRE,
                                     __HIP_MEMORY_SCOPE_AGENT) < target)
                __builtin_amdgcn_s_sleep(1);
        }
        __syncthreads();

        // ---- broadcast: reload full x_n (8 KB, L2-hot) into LDS as bf16 ----
        if (n + 1 < T_STEPS) {
            const float* xsrc = states + (size_t)n * R;
            for (int i = tid; i < R; i += BLOCK) xb[i] = (__bf16)xsrc[i];
            __syncthreads();
        }
    }
}

// out[t] = states[t] @ lin_w[:, :2048]^T + (t>0 ? u_t @ lin_w[:, 2048:]^T : 0) + b
__global__ void readout_kernel(const float* __restrict__ states,
                               const float* __restrict__ inputs,
                               const float* __restrict__ lin_w,   // [4, 2056]
                               const float* __restrict__ lin_b,   // [4]
                               float*       __restrict__ out)     // [T, 4]
{
    __shared__ float red[256 * N_OUT];
    const int t   = blockIdx.x;
    const int tid = threadIdx.x;

    float acc[N_OUT] = {0.f, 0.f, 0.f, 0.f};
    const float* srow = states + (size_t)t * R;
    for (int r = tid; r < R; r += 256) {
        float s = srow[r];
        #pragma unroll
        for (int o = 0; o < N_OUT; ++o)
            acc[o] += s * lin_w[(size_t)o * (R + N_IN) + r];
    }
    #pragma unroll
    for (int o = 0; o < N_OUT; ++o) red[tid * N_OUT + o] = acc[o];
    __syncthreads();
    for (int s = 128; s > 0; s >>= 1) {
        if (tid < s) {
            #pragma unroll
            for (int o = 0; o < N_OUT; ++o)
                red[tid * N_OUT + o] += red[(tid + s) * N_OUT + o];
        }
        __syncthreads();
    }
    if (tid < N_OUT) {
        float v = red[tid] + lin_b[tid];
        if (t > 0) {   // states_u row 0 is zero in the reference
            #pragma unroll
            for (int i = 0; i < N_IN; ++i)
                v += inputs[(size_t)t * N_IN + i] *
                     lin_w[(size_t)tid * (R + N_IN) + R + i];
        }
        out[(size_t)t * N_OUT + tid] = v;
    }
}

extern "C" void kernel_launch(void* const* d_in, const int* in_sizes, int n_in,
                              void* d_out, int out_size, void* d_ws, size_t ws_size,
                              hipStream_t stream)
{
    const float* inputs  = (const float*)d_in[0];
    const float* outputs = (const float*)d_in[1];
    const float* w       = (const float*)d_in[2];
    const float* w_in    = (const float*)d_in[3];
    const float* w_feedb = (const float*)d_in[4];
    const float* lin_w   = (const float*)d_in[5];
    const float* lin_b   = (const float*)d_in[6];
    const float* noise   = (const float*)d_in[7];

    unsigned* cnt    = (unsigned*)d_ws;                 // barrier counter @ 0
    float*    states = (float*)((char*)d_ws + 256);     // [T, R] f32 (33.6 MB)

    // reset barrier counter every call (capture-safe memset node)
    hipMemsetAsync(d_ws, 0, 256, stream);

    // allow > 64 KB dynamic LDS (CDNA5: 320 KB/WGP); idempotent, capture-safe
    hipFuncSetAttribute((const void*)reservoir_kernel,
                        hipFuncAttributeMaxDynamicSharedMemorySize,
                        (int)LDS_BYTES);

    reservoir_kernel<<<NWG, BLOCK, LDS_BYTES, stream>>>(
        inputs, outputs, w, w_in, w_feedb, noise, states, cnt);

    readout_kernel<<<T_STEPS, 256, 0, stream>>>(
        states, inputs, lin_w, lin_b, (float*)d_out);
}
